// MultiheadAttention_1537598292749
// MI455X (gfx1250) — compile-verified, matrix operand order checked by reading, not run on hardware
//
#include <hip/hip_runtime.h>
#include <hip/hip_bf16.h>

// ---- problem constants -------------------------------------------------
#define S_LEN 2048
#define B_SZ  4
#define E_DIM 1024
#define H_NUM 16
#define D_DIM 64
#define M_ROWS (S_LEN * B_SZ)   // 8192
#define N3     (3 * E_DIM)      // 3072
#define NHEAD  (B_SZ * H_NUM)   // 64

// ---- WMMA types --------------------------------------------------------
typedef __bf16 bf16_t;
typedef __attribute__((ext_vector_type(16))) __bf16 v16bf;
typedef __attribute__((ext_vector_type(8)))  float  v8f;
typedef __attribute__((ext_vector_type(4)))  unsigned int v4u;

union Frag16 {          // 16 bf16 elements = 8 VGPRs, loaded as two 16B strips
    v16bf v;
    v4u   q[2];
};

#define WMMA_BF16(A, B, C) \
    __builtin_amdgcn_wmma_f32_16x16x32_bf16(false, (A), false, (B), (short)0, (C), false, false)

// A/B fragment addressing (wave32, 16x16x32 bf16):
//   lane 0-15  : row/col = lane,    K strip0 = [0..7],  strip1 = [16..23]
//   lane 16-31 : row/col = lane-16, K strip0 = [8..15], strip1 = [24..31]
// C/D f32: slot j, lanes 0-15 -> (M=j, N=lane); lanes 16-31 -> (M=j+8, N=lane-16)

// ---- fp32 -> bf16 conversion ------------------------------------------
__global__ void k_cvt_bf16(const float* __restrict__ src, bf16_t* __restrict__ dst, int n) {
    for (int i = blockIdx.x * blockDim.x + threadIdx.x; i < n; i += gridDim.x * blockDim.x)
        dst[i] = (bf16_t)src[i];
}

// ---- kernel 1: QKV projection  qkv = X @ Win^T + bias ------------------
// Register-blocked: each wave computes a 32x64 output tile (8 WMMAs / K-step,
// 12 b128 loads / 8 WMMAs). X (8192x1024) bf16, Win (3072x1024) bf16.
// Writes: q (head,S,D) * (D^-0.5 * log2 e) ; k (head,S,D) ; vT (head,D,S)
__global__ void k_qkv(const bf16_t* __restrict__ X, const bf16_t* __restrict__ W,
                      const float* __restrict__ bias,
                      bf16_t* __restrict__ qb, bf16_t* __restrict__ kb,
                      bf16_t* __restrict__ vT) {
    const int wave = (blockIdx.x * blockDim.x + threadIdx.x) >> 5;
    const int lane = threadIdx.x & 31;
    const int tm = wave / (N3 / 64);           // 0..255 (32-row tiles)
    const int tn = wave % (N3 / 64);           // 0..47  (64-col tiles)
    const int row0 = tm * 32, col0 = tn * 64;
    const int r    = lane & 15;
    const int koff = (lane >> 4) * 8;

    v8f acc[2][4] = {};
    const bf16_t* ax0 = X + (size_t)(row0 + r) * E_DIM + koff;
    const bf16_t* ax1 = ax0 + 16 * E_DIM;
    const bf16_t* bw  = W + (size_t)(col0 + r) * E_DIM + koff;

#pragma unroll 4
    for (int kk = 0; kk < E_DIM; kk += 32) {
        Frag16 a0, a1;
        a0.q[0] = *(const v4u*)(ax0 + kk); a0.q[1] = *(const v4u*)(ax0 + kk + 16);
        a1.q[0] = *(const v4u*)(ax1 + kk); a1.q[1] = *(const v4u*)(ax1 + kk + 16);
#pragma unroll
        for (int t = 0; t < 4; ++t) {
            Frag16 b;
            const bf16_t* bp = bw + (size_t)t * 16 * E_DIM + kk;
            b.q[0] = *(const v4u*)(bp); b.q[1] = *(const v4u*)(bp + 16);
            acc[0][t] = WMMA_BF16(a0.v, b.v, acc[0][t]);
            acc[1][t] = WMMA_BF16(a1.v, b.v, acc[1][t]);
        }
    }

    // q pre-scale: D^-0.5 * log2(e) so attention softmax can run in exp2
    const float qscale = 0.125f * 1.44269504f;
    const int rb = (lane >> 4) * 8;

#pragma unroll
    for (int t = 0; t < 4; ++t) {
        const int col = col0 + t * 16 + (lane & 15);
        const float bf = bias[col];
        const int sec = col >> 10;             // 0=q 1=k 2=v (uniform per tile)
        const int e   = col & (E_DIM - 1);
        const int h   = e >> 6;
        const int d   = e & (D_DIM - 1);
#pragma unroll
        for (int half = 0; half < 2; ++half) {
#pragma unroll
            for (int j = 0; j < 8; ++j) {
                const int m  = row0 + half * 16 + rb + j;   // row = s*B + b
                const int s  = m >> 2;
                const int bb = m & 3;
                const int n  = bb * H_NUM + h;
                const float val = acc[half][t][j] + bf;
                if (sec == 0)      qb[((size_t)n * S_LEN + s) * D_DIM + d] = (bf16_t)(val * qscale);
                else if (sec == 1) kb[((size_t)n * S_LEN + s) * D_DIM + d] = (bf16_t)val;
                else               vT[((size_t)n * D_DIM + d) * S_LEN + s] = (bf16_t)val;
            }
        }
    }
}

// ---- kernel 2: flash attention, one wave per (head, 16-query block) ----
// 64-key chunks: softmax reductions/rescales amortized 2x vs 32-key chunks.
#define P_STRIDE 72      // halves; 144B rows -> 16B aligned, conflict-free
__global__ void k_attn(const bf16_t* __restrict__ qb, const bf16_t* __restrict__ kb,
                       const bf16_t* __restrict__ vT, bf16_t* __restrict__ obuf) {
    __shared__ bf16_t ldsP[8 * 16 * P_STRIDE];

    const int wave  = (blockIdx.x * blockDim.x + threadIdx.x) >> 5;
    const int lane  = threadIdx.x & 31;
    const int wslot = (threadIdx.x >> 5) & 7;
    const int head  = wave >> 7;               // / 128
    const int qblk  = wave & 127;
    const int q0    = qblk * 16;
    const int r     = lane & 15;
    const int koff  = (lane >> 4) * 8;

    // Q fragments (row q0+r, D=64 -> two K=32 fragments), resident in VGPRs
    Frag16 qa0, qa1;
    {
        const bf16_t* qp = qb + ((size_t)head * S_LEN + q0 + r) * D_DIM + koff;
        qa0.q[0] = *(const v4u*)(qp);       qa0.q[1] = *(const v4u*)(qp + 16);
        qa1.q[0] = *(const v4u*)(qp + 32);  qa1.q[1] = *(const v4u*)(qp + 48);
    }

    v8f o[4] = {};
    float mrow[8], lrow[8];
#pragma unroll
    for (int j = 0; j < 8; ++j) { mrow[j] = -3.0e38f; lrow[j] = 0.0f; }

    bf16_t* myP = ldsP + wslot * 16 * P_STRIDE;
    const bf16_t* kbase = kb + ((size_t)head * S_LEN + r) * D_DIM + koff;
    const bf16_t* vbase = vT + ((size_t)head * D_DIM + r) * S_LEN + koff;

    for (int kc = 0; kc < S_LEN; kc += 64) {
        const bf16_t* kp = kbase + (size_t)kc * D_DIM;
        const bf16_t* vp = vbase + kc;

        // prefetch next chunk's K/V streams into cache (global_prefetch_b8)
        if (kc + 64 < S_LEN) {
            __builtin_prefetch(kp + 64 * D_DIM, 0, 0);
            __builtin_prefetch(vp + 64, 0, 0);
        }

        // ---- scores: four 16x16 tiles (keys kc+16t .. kc+16t+15) -------
        v8f s[4] = {};
#pragma unroll
        for (int t = 0; t < 4; ++t) {
            const bf16_t* kt = kp + (size_t)t * 16 * D_DIM;
            Frag16 b;
            b.q[0] = *(const v4u*)(kt);      b.q[1] = *(const v4u*)(kt + 16);
            s[t] = WMMA_BF16(qa0.v, b.v, s[t]);
            b.q[0] = *(const v4u*)(kt + 32); b.q[1] = *(const v4u*)(kt + 48);
            s[t] = WMMA_BF16(qa1.v, b.v, s[t]);
        }

        // ---- online softmax in base-2 (scores pre-scaled by log2 e) ----
#pragma unroll
        for (int j = 0; j < 8; ++j) {
            float mv = fmaxf(fmaxf(s[0][j], s[1][j]), fmaxf(s[2][j], s[3][j]));
            mv = fmaxf(mv, __shfl_xor(mv, 1));
            mv = fmaxf(mv, __shfl_xor(mv, 2));
            mv = fmaxf(mv, __shfl_xor(mv, 4));
            mv = fmaxf(mv, __shfl_xor(mv, 8));
            const float mnew = fmaxf(mrow[j], mv);
            const float corr = exp2f(mrow[j] - mnew);
            float p[4];
#pragma unroll
            for (int t = 0; t < 4; ++t) p[t] = exp2f(s[t][j] - mnew);
            float ps = (p[0] + p[1]) + (p[2] + p[3]);
            ps += __shfl_xor(ps, 1);
            ps += __shfl_xor(ps, 2);
            ps += __shfl_xor(ps, 4);
            ps += __shfl_xor(ps, 8);
            lrow[j] = lrow[j] * corr + ps;
            mrow[j] = mnew;
#pragma unroll
            for (int t = 0; t < 4; ++t) o[t][j] *= corr;
            const int prow = j + ((lane >> 4) << 3);
#pragma unroll
            for (int t = 0; t < 4; ++t)
                myP[prow * P_STRIDE + t * 16 + (lane & 15)] = (bf16_t)p[t];
        }

        // ---- reload P as two A-fragments (keys 0..31, 32..63 of chunk) -
        Frag16 pa0, pa1;
        {
            const bf16_t* pp = myP + r * P_STRIDE + koff;
            pa0.q[0] = *(const v4u*)(pp);      pa0.q[1] = *(const v4u*)(pp + 16);
            pa1.q[0] = *(const v4u*)(pp + 32); pa1.q[1] = *(const v4u*)(pp + 48);
        }

        // ---- out += P @ V  (4 N-tiles across D=64, 2 K-steps each) -----
#pragma unroll
        for (int t = 0; t < 4; ++t) {
            const bf16_t* vt = vp + (size_t)t * 16 * S_LEN;
            Frag16 b;
            b.q[0] = *(const v4u*)(vt);      b.q[1] = *(const v4u*)(vt + 16);
            o[t] = WMMA_BF16(pa0.v, b.v, o[t]);
            b.q[0] = *(const v4u*)(vt + 32); b.q[1] = *(const v4u*)(vt + 48);
            o[t] = WMMA_BF16(pa1.v, b.v, o[t]);
        }
    }

    // ---- normalize and store to (S,B,E) bf16 buffer ---------------------
    const int bb = head >> 4;                  // n = b*H + h
    const int hh = head & 15;
    const int rb = (lane >> 4) << 3;
#pragma unroll
    for (int j = 0; j < 8; ++j) {
        const float rinv = 1.0f / lrow[j];
        const int s = q0 + rb + j;
        const size_t base = ((size_t)s * B_SZ + bb) * E_DIM + hh * D_DIM + (lane & 15);
#pragma unroll
        for (int t = 0; t < 4; ++t)
            obuf[base + t * 16] = (bf16_t)(o[t][j] * rinv);
    }
}

// ---- kernel 3: output projection  out = A @ Wout^T + bias (fp32 out) ---
// Register-blocked 32x64 tile per wave, same scheme as k_qkv.
__global__ void k_proj(const bf16_t* __restrict__ A, const bf16_t* __restrict__ W,
                       const float* __restrict__ bias, float* __restrict__ out) {
    const int wave = (blockIdx.x * blockDim.x + threadIdx.x) >> 5;
    const int lane = threadIdx.x & 31;
    const int tm = wave / (E_DIM / 64);        // 0..255
    const int tn = wave % (E_DIM / 64);        // 0..15
    const int row0 = tm * 32, col0 = tn * 64;
    const int r    = lane & 15;
    const int koff = (lane >> 4) * 8;

    v8f acc[2][4] = {};
    const bf16_t* ax0 = A + (size_t)(row0 + r) * E_DIM + koff;
    const bf16_t* ax1 = ax0 + 16 * E_DIM;
    const bf16_t* bw  = W + (size_t)(col0 + r) * E_DIM + koff;

#pragma unroll 4
    for (int kk = 0; kk < E_DIM; kk += 32) {
        Frag16 a0, a1;
        a0.q[0] = *(const v4u*)(ax0 + kk); a0.q[1] = *(const v4u*)(ax0 + kk + 16);
        a1.q[0] = *(const v4u*)(ax1 + kk); a1.q[1] = *(const v4u*)(ax1 + kk + 16);
#pragma unroll
        for (int t = 0; t < 4; ++t) {
            Frag16 b;
            const bf16_t* bp = bw + (size_t)t * 16 * E_DIM + kk;
            b.q[0] = *(const v4u*)(bp); b.q[1] = *(const v4u*)(bp + 16);
            acc[0][t] = WMMA_BF16(a0.v, b.v, acc[0][t]);
            acc[1][t] = WMMA_BF16(a1.v, b.v, acc[1][t]);
        }
    }

    const int rb = (lane >> 4) * 8;
#pragma unroll
    for (int t = 0; t < 4; ++t) {
        const int col = col0 + t * 16 + (lane & 15);
        const float bf = bias[col];
#pragma unroll
        for (int half = 0; half < 2; ++half) {
#pragma unroll
            for (int j = 0; j < 8; ++j)
                out[(size_t)(row0 + half * 16 + rb + j) * E_DIM + col] = acc[half][t][j] + bf;
        }
    }
}

// ---- host launch --------------------------------------------------------
extern "C" void kernel_launch(void* const* d_in, const int* in_sizes, int n_in,
                              void* d_out, int out_size, void* d_ws, size_t ws_size,
                              hipStream_t stream) {
    const float* query = (const float*)d_in[0];   // (S,B,E)
    const float* Win   = (const float*)d_in[1];   // (3E,E)
    const float* bin   = (const float*)d_in[2];   // (3E)
    const float* Wout  = (const float*)d_in[3];   // (E,E)
    const float* bout  = (const float*)d_in[4];   // (E)
    float* out = (float*)d_out;

    char* ws = (char*)d_ws;
    bf16_t* Xbf   = (bf16_t*)(ws);                               // 16 MB
    bf16_t* Winb  = (bf16_t*)(ws + (16u << 20));                 //  6 MB
    bf16_t* Woutb = (bf16_t*)(ws + (16u << 20) + (6u << 20));    //  2 MB
    bf16_t* qbuf  = (bf16_t*)(ws + (24u << 20));                 // 16 MB
    bf16_t* kbuf  = (bf16_t*)(ws + (40u << 20));                 // 16 MB
    bf16_t* vTbuf = (bf16_t*)(ws + (56u << 20));                 // 16 MB
    bf16_t* obuf  = (bf16_t*)(ws + (72u << 20));                 // 16 MB

    // fp32 -> bf16
    k_cvt_bf16<<<4096, 256, 0, stream>>>(query, Xbf,  M_ROWS * E_DIM);
    k_cvt_bf16<<<2048, 256, 0, stream>>>(Win,   Winb, N3 * E_DIM);
    k_cvt_bf16<<<1024, 256, 0, stream>>>(Wout,  Woutb, E_DIM * E_DIM);

    // QKV projection: 256 x 48 tiles of 32x64, 1 wave each, 8 waves/block
    k_qkv<<<(256 * 48) / 8, 256, 0, stream>>>(Xbf, Winb, bin, qbuf, kbuf, vTbuf);

    // flash attention: 64 heads x 128 query blocks = 8192 waves
    k_attn<<<8192 / 8, 256, 0, stream>>>(qbuf, kbuf, vTbuf, obuf);

    // output projection: 256 x 16 tiles of 32x64
    k_proj<<<(256 * 16) / 8, 256, 0, stream>>>(obuf, Woutb, bout, out);
}
